// AdditiveAttention_54692113548010
// MI455X (gfx1250) — matrix operand order repaired and checked by reading, not compiled, not run on hardware
//
#include <hip/hip_runtime.h>
#include <hip/hip_bf16.h>
#include <stdint.h>

#define USE_ASYNC_LDS 1
#define USE_TDM 1

#define B_ 16
#define Q_ 256
#define K_ 256
#define D_ 256
#define H_ 256

typedef __attribute__((ext_vector_type(16))) __bf16 v16bf;
typedef __attribute__((ext_vector_type(8)))  float  v8f;
typedef unsigned int u32x4 __attribute__((ext_vector_type(4)));
typedef int          i32x8 __attribute__((ext_vector_type(8)));
typedef int          i32x4 __attribute__((ext_vector_type(4)));

// ---- tanh: force the CDNA5 TRANS-pipe instruction --------------------------
#if __has_builtin(__builtin_amdgcn_tanhf)
#define TANHF(x) __builtin_amdgcn_tanhf(x)
#elif __has_builtin(__builtin_amdgcn_tanh_f32)
#define TANHF(x) __builtin_amdgcn_tanh_f32(x)
#else
__device__ __forceinline__ float __tanh_asm(float x) {
  float r;
  asm volatile("v_tanh_f32 %0, %1" : "=v"(r) : "v"(x));
  return r;
}
#define TANHF(x) __tanh_asm(x)
#endif

// ---- CDNA5 async global->LDS copy (ASYNCcnt DMA path) ----------------------
__device__ __forceinline__ void async_copy_b128(void* lds_dst, const void* gsrc) {
#if USE_ASYNC_LDS
  unsigned lds = (unsigned)(uintptr_t)lds_dst;   // low 32 bits = LDS byte offset
  asm volatile("global_load_async_to_lds_b128 %0, %1, off"
               :: "v"(lds), "v"(gsrc) : "memory");
#else
  *(float4*)lds_dst = *(const float4*)gsrc;
#endif
}
__device__ __forceinline__ void async_wait_all() {
#if USE_ASYNC_LDS
  asm volatile("s_wait_asynccnt 0x0" ::: "memory");
#endif
}

// ---- CDNA5 Tensor Data Mover: 2D tile [rows x 256 f32] -> LDS --------------
// Tensor in memory: [4096 x 256] f32, row stride 256 elements.
// LDS destination rows padded to 260 dwords via D# pad fields
// (pad_interval=7 -> every 256 dwords, pad_amount=3 -> 4 dwords).
__device__ __forceinline__ void tdm_load_rows_padded(void* lds_dst,
                                                     const void* gsrc,
                                                     int tile_rows) {
#if USE_TDM
  unsigned lds = (unsigned)(uintptr_t)lds_dst;
  unsigned long long ga = (unsigned long long)(uintptr_t)gsrc;
  u32x4 g0;
  g0[0] = 1u;                                      // count=1 (valid), user mode
  g0[1] = lds;                                     // lds_addr
  g0[2] = (unsigned)(ga & 0xFFFFFFFFu);            // global_addr[31:0]
  g0[3] = (unsigned)((ga >> 32) & 0x01FFFFFFu)     // global_addr[56:32]
        | 0x80000000u;                             // type = 2 ("image")
  i32x8 g1;
  g1[0] = (2 << 16)                                // data_size = 4 bytes
        | (1 << 20)                                // pad_enable
        | (7 << 22)                                // pad_interval: 256 dwords
        | (3 << 25);                               // pad_amount: 4 dwords
  g1[1] = (H_ & 0xFFFF) << 16;                     // tensor_dim0 = 256 (lo16)
  g1[2] = ((B_ * K_) & 0xFFFF) << 16;              // dim0 hi=0 | tensor_dim1 = 4096 (lo16)
  g1[3] = (H_ << 16);                              // dim1 hi=0 | tile_dim0 = 256
  g1[4] = tile_rows & 0xFFFF;                      // tile_dim1 = rows, tile_dim2 = 0
  g1[5] = H_;                                      // tensor_dim0_stride = 256
  g1[6] = 0;
  g1[7] = 0;
  i32x4 z4 = {0, 0, 0, 0};                         // 2D: groups 2/3 unused
  i32x8 z8 = {0, 0, 0, 0, 0, 0, 0, 0};
  __builtin_amdgcn_tensor_load_to_lds(g0, g1, z4, z4, z8, 0);
  __builtin_amdgcn_s_wait_tensorcnt(0);
#else
  (void)lds_dst; (void)gsrc; (void)tile_rows;
#endif
}

// ---------------------------------------------------------------------------
// Kernel 1: Y[r,h] = sum_d X[r,d] * W[d,h]   (rows = B*Q = 4096, D=H=256)
// One wave per 16x16 output tile; 8 x v_wmma_f32_16x16x32_bf16 along D.
// ---------------------------------------------------------------------------
__global__ void __launch_bounds__(256) proj_wmma_kernel(const float* __restrict__ X,
                                                        const float* __restrict__ W,
                                                        float* __restrict__ Y) {
  const int lane  = threadIdx.x & 31;
  const int wave  = blockIdx.x * 8 + (threadIdx.x >> 5);  // 4096 waves total
  const int tileM = wave >> 4;                            // 0..255
  const int tileN = wave & 15;                            // 0..15
  const int ln15  = lane & 15;
  const int kh    = lane >> 4;                            // lane half: 0/1

  const float* arow = X + (size_t)(tileM * 16 + ln15) * D_;
  v8f acc = {};
#pragma unroll
  for (int dc = 0; dc < D_ / 32; ++dc) {
    v16bf a, b;
#pragma unroll
    for (int v = 0; v < 8; ++v) {
      // A, contiguous k = 16*half + s (B matches): VGPR v holds a K pair; lanes 16-31 hold K 16-31
      int ka = (kh << 4) + 2 * v;
      a[2 * v]     = (__bf16)arow[dc * 32 + ka];
      a[2 * v + 1] = (__bf16)arow[dc * 32 + ka + 1];
      // B 16-bit 32x16 layout: lane = N col; VGPR v holds a K pair
      int kb = (kh << 4) + 2 * v;
      const float* wp = W + (size_t)(dc * 32 + kb) * H_ + tileN * 16 + ln15;
      b[2 * v]     = (__bf16)wp[0];
      b[2 * v + 1] = (__bf16)wp[H_];
    }
    acc = __builtin_amdgcn_wmma_f32_16x16x32_bf16(false, a, false, b,
                                                  (short)0, acc, false, false);
  }
  // C layout: VGPR i -> row i (lanes 0-15) / row 8+i (lanes 16-31)
  float* yb = Y + (size_t)(tileM * 16) * H_ + tileN * 16 + ln15;
#pragma unroll
  for (int i = 0; i < 8; ++i) {
    int mm = (kh << 3) + i;
    yb[(size_t)mm * H_] = acc[i];
  }
}

// ---------------------------------------------------------------------------
// Kernel 2: fused scores(tanh) + masked softmax + attn@V.
// One 256-thread WG (8 waves) per (batch b, 16-row q tile). L2-resident data.
// ---------------------------------------------------------------------------
__global__ void __launch_bounds__(256) attn_fused_kernel(
    const float* __restrict__ qW, const float* __restrict__ kW,
    const float* __restrict__ V, const long long* __restrict__ valid_lens,
    const float* __restrict__ wv, float* __restrict__ out) {

  __shared__ __align__(16) float sQ[16 * 260];  // 260-dword rows: conflict-free
  __shared__ __align__(16) float sK[16 * 260];
  __shared__ __align__(16) float sS[16 * 260];  // scores -> attn weights
  __shared__ __align__(16) float sV[32 * 260];
  __shared__ __align__(16) float sWv[H_];
  __shared__ float sRed[256];
  __shared__ float sRowMax[16];
  __shared__ float sRowSum[16];

  const int t    = threadIdx.x;
  const int b    = blockIdx.x >> 4;
  const int qt   = blockIdx.x & 15;
  const int qrow = t & 15;
  const int kcol = t >> 4;

  const float* qWbase = qW + ((size_t)b * Q_ + qt * 16) * H_;
  const float* kWbase = kW + (size_t)b * K_ * H_;
  const float* Vbase  = V  + (size_t)b * K_ * D_;
  const int vlen = (int)valid_lens[b];

  // ---- Phase 1: stage q tile + wv via async-LDS DMA ----
  for (int c = t; c < 16 * 64; c += 256) {
    int r = c >> 6, j = (c & 63) << 2;
    async_copy_b128(&sQ[r * 260 + j], qWbase + (size_t)r * H_ + j);
  }
  if (t < 64) async_copy_b128(&sWv[t << 2], wv + (t << 2));
  async_wait_all();
  __syncthreads();

  // ---- Phase 2: scores (the transcendental roofline term) ----
  for (int kt = 0; kt < 16; ++kt) {
    __syncthreads();                          // sK consumed by previous iter
    if (t < 32)                               // wave 0 drives the TDM
      tdm_load_rows_padded(&sK[0], kWbase + (size_t)(kt * 16) * H_, 16);
    if (kt + 1 < 16)
      __builtin_prefetch(kWbase + (size_t)((kt + 1) * 16 + qrow) * H_, 0, 1);
    __syncthreads();

    const float* qr = &sQ[qrow * 260];
    const float* kr = &sK[kcol * 260];
    float acc = 0.f;
#pragma unroll 8
    for (int h = 0; h < H_; ++h)
      acc += sWv[h] * TANHF(qr[h] + kr[h]);
    sS[qrow * 260 + kt * 16 + kcol] = acc;
  }
  __syncthreads();

  // ---- Phase 3: mask (-1e6 like reference) + softmax over k ----
  {
    float lmax = -3.0e38f;
#pragma unroll
    for (int j = 0; j < 16; ++j) {
      int k = kcol * 16 + j;
      float s = sS[qrow * 260 + k];
      if (k >= vlen) s = -1.0e6f;
      sS[qrow * 260 + k] = s;
      lmax = fmaxf(lmax, s);
    }
    sRed[t] = lmax;
    __syncthreads();
    if (t < 16) {
      float m = -3.0e38f;
      for (int sg = 0; sg < 16; ++sg) m = fmaxf(m, sRed[sg * 16 + t]);
      sRowMax[t] = m;
    }
    __syncthreads();
    float rmax = sRowMax[qrow];
    float lsum = 0.f;
#pragma unroll
    for (int j = 0; j < 16; ++j) {
      int k = kcol * 16 + j;
      float e = __expf(sS[qrow * 260 + k] - rmax);
      sS[qrow * 260 + k] = e;
      lsum += e;
    }
    sRed[t] = lsum;
    __syncthreads();
    if (t < 16) {
      float ssum = 0.f;
      for (int sg = 0; sg < 16; ++sg) ssum += sRed[sg * 16 + t];
      sRowSum[t] = ssum;
    }
    __syncthreads();
    float inv = 1.0f / sRowSum[qrow];
#pragma unroll
    for (int j = 0; j < 16; ++j)
      sS[qrow * 260 + kcol * 16 + j] *= inv;
  }

  // ---- Phase 4: out = attn @ V, bf16 WMMA, f32 accumulate ----
  const int lane = t & 31;
  const int wv8  = t >> 5;          // wave 0..7 owns d-tiles wv8 and wv8+8
  const int ln15 = lane & 15;
  const int kh   = lane >> 4;
  const int dt0 = wv8, dt1 = wv8 + 8;
  v8f acc0 = {}, acc1 = {};

  for (int kc = 0; kc < 8; ++kc) {
    __syncthreads();                 // previous sV fully consumed
    if (t < 32)                      // wave 0 drives the TDM (32 rows of V)
      tdm_load_rows_padded(&sV[0], Vbase + (size_t)(kc * 32) * D_, 32);
    __syncthreads();

    v16bf a, b0, b1;
#pragma unroll
    for (int v = 0; v < 8; ++v) {
      int ka = (kh << 4) + 2 * v;
      a[2 * v]     = (__bf16)sS[ln15 * 260 + kc * 32 + ka];
      a[2 * v + 1] = (__bf16)sS[ln15 * 260 + kc * 32 + ka + 1];
      int kb = (kh << 4) + 2 * v;
      b0[2 * v]     = (__bf16)sV[kb * 260 + dt0 * 16 + ln15];
      b0[2 * v + 1] = (__bf16)sV[(kb + 1) * 260 + dt0 * 16 + ln15];
      b1[2 * v]     = (__bf16)sV[kb * 260 + dt1 * 16 + ln15];
      b1[2 * v + 1] = (__bf16)sV[(kb + 1) * 260 + dt1 * 16 + ln15];
    }
    acc0 = __builtin_amdgcn_wmma_f32_16x16x32_bf16(false, a, false, b0,
                                                   (short)0, acc0, false, false);
    acc1 = __builtin_amdgcn_wmma_f32_16x16x32_bf16(false, a, false, b1,
                                                   (short)0, acc1, false, false);
  }

  float* ob = out + ((size_t)b * Q_ + qt * 16) * D_;
#pragma unroll
  for (int i = 0; i < 8; ++i) {
    int mm = (kh << 3) + i;
    ob[(size_t)mm * D_ + dt0 * 16 + ln15] = acc0[i];
    ob[(size_t)mm * D_ + dt1 * 16 + ln15] = acc1[i];
  }
}

// ---------------------------------------------------------------------------
extern "C" void kernel_launch(void* const* d_in, const int* in_sizes, int n_in,
                              void* d_out, int out_size, void* d_ws, size_t ws_size,
                              hipStream_t stream) {
  const float*     queries = (const float*)d_in[0];
  const float*     keys    = (const float*)d_in[1];
  const float*     values  = (const float*)d_in[2];
  const long long* vlens   = (const long long*)d_in[3];  // int64 in reference
  const float*     Wq      = (const float*)d_in[4];
  const float*     Wk      = (const float*)d_in[5];
  const float*     wv      = (const float*)d_in[6];
  float* out = (float*)d_out;

  float* qW = (float*)d_ws;                      // [B*Q, H] = 4 MB
  float* kW = qW + (size_t)B_ * Q_ * H_;         // [B*K, H] = 4 MB

  // 256 row-tiles x 16 col-tiles = 4096 waves = 512 blocks of 8 waves
  proj_wmma_kernel<<<512, 256, 0, stream>>>(queries, Wq, qW);
  proj_wmma_kernel<<<512, 256, 0, stream>>>(keys,    Wk, kW);
  attn_fused_kernel<<<B_ * (Q_ / 16), 256, 0, stream>>>(qW, kW, values, vlens, wv, out);

  (void)in_sizes; (void)n_in; (void)out_size; (void)ws_size;
}
